// CausalSelfAttention_1571958030818
// MI455X (gfx1250) — matrix operand order repaired and checked by reading, not compiled, not run on hardware
//
#include <hip/hip_runtime.h>
#include <hip/hip_bf16.h>

// Problem constants (from reference): B=4, T=2048, C=1024, H=16, D=64
#define B_  4
#define T_  2048
#define C_  1024
#define H_  16
#define D_  64
#define C3_ 3072

typedef __attribute__((ext_vector_type(16))) __bf16       v16bf;
typedef __attribute__((ext_vector_type(8)))  float        v8f;
typedef __attribute__((ext_vector_type(4)))  unsigned int v4u;

union Frag { v16bf v; v4u u[2]; };

__device__ __forceinline__ v8f wmma_bf16(v16bf a, v16bf b, v8f c) {
  // D = A(16x32 bf16) * B(32x16 bf16) + C(16x16 f32)
  return __builtin_amdgcn_wmma_f32_16x16x32_bf16(
      false, a, false, b, (short)0, c, false, false);
}

// A-operand (16x32, 16-bit) per ISA 7.12.2:
// lanes 0-15 hold K {0..7} in V0-3 and {16..23} in V4-7 (row m = lane&15),
// lanes 16-31 hold K {8..15} and {24..31}.  koffA = (lane>>4)*8.
__device__ __forceinline__ void load_a(Frag& f, const __bf16* rowptr, int kk, int koffA) {
  f.u[0] = *(const v4u*)(rowptr + kk + koffA);
  f.u[1] = *(const v4u*)(rowptr + kk + koffA + 16);
}
// B-operand (32x16, 16-bit): column n = lane&15, same K order as A:
// lanes 0-15 hold K {0..7},{16..23}, lanes 16-31 hold K {8..15},{24..31}.  koffB = (lane>>4)*8.
// Source is stored transposed ([N][K] row-major) so the column is contiguous.
__device__ __forceinline__ void load_b(Frag& f, const __bf16* rowptr, int kk, int koffB) {
  f.u[0] = *(const v4u*)(rowptr + kk + koffB);
  f.u[1] = *(const v4u*)(rowptr + kk + koffB + 16);
}

// ---------------- conversion / transpose kernels ----------------
__global__ void k_cvt(const float* __restrict__ in, __bf16* __restrict__ out, int n) {
  int i = blockIdx.x * blockDim.x + threadIdx.x;
  int stride = gridDim.x * blockDim.x;
  for (; i < n; i += stride) out[i] = (__bf16)in[i];
}

// in [rows][cols] f32  ->  out [cols][rows] bf16
__global__ void k_transpose(const float* __restrict__ in, __bf16* __restrict__ out,
                            int rows, int cols) {
  size_t n = (size_t)rows * cols;
  size_t stride = (size_t)gridDim.x * blockDim.x;
  for (size_t i = blockIdx.x * (size_t)blockDim.x + threadIdx.x; i < n; i += stride) {
    int r = (int)(i / cols), c = (int)(i % cols);
    out[(size_t)c * rows + r] = (__bf16)in[i];
  }
}

// ---------------- QKV GEMM: qkv = Xb[8192,1024] @ WqkvT^T + bias ----------------
// 8 waves/block; wave tile = 32x64 (2x4 WMMA tiles). Epilogue scatters to
// Q,K as [B,H,T,D] bf16 and V as [B,H,D,T] bf16 (pre-transposed for PV GEMM).
__global__ __launch_bounds__(256) void k_gemm_qkv(
    const __bf16* __restrict__ Xb, const __bf16* __restrict__ Wt,
    const float* __restrict__ bias,
    __bf16* __restrict__ Qb, __bf16* __restrict__ Kb, __bf16* __restrict__ Vt) {
  const int tid = threadIdx.x;
  const int lane = tid & 31, wv = tid >> 5;
  const int half = lane >> 4, l15 = lane & 15;
  const int koffA = half * 8, koffB = half * 8;
  const int m0 = blockIdx.y * 64 + (wv >> 2) * 32;
  const int n0 = blockIdx.x * 256 + (wv & 3) * 64;

  const v8f vzero = {0.f,0.f,0.f,0.f,0.f,0.f,0.f,0.f};
  v8f acc[2][4];
  for (int i = 0; i < 2; ++i)
    for (int j = 0; j < 4; ++j) acc[i][j] = vzero;

  const __bf16* arow0 = Xb + (size_t)(m0 + l15) * C_;
  const __bf16* arow1 = Xb + (size_t)(m0 + 16 + l15) * C_;
  const __bf16* brow[4];
  for (int j = 0; j < 4; ++j) brow[j] = Wt + (size_t)(n0 + 16 * j + l15) * C_;

  for (int kk = 0; kk < C_; kk += 32) {
    Frag a0, a1, bf[4];
    load_a(a0, arow0, kk, koffA);
    load_a(a1, arow1, kk, koffA);
    for (int j = 0; j < 4; ++j) load_b(bf[j], brow[j], kk, koffB);
    for (int j = 0; j < 4; ++j) {
      acc[0][j] = wmma_bf16(a0.v, bf[j].v, acc[0][j]);
      acc[1][j] = wmma_bf16(a1.v, bf[j].v, acc[1][j]);
    }
  }

  for (int mi = 0; mi < 2; ++mi)
    for (int ni = 0; ni < 4; ++ni)
      for (int r = 0; r < 8; ++r) {
        int m = m0 + 16 * mi + r + 8 * half;     // C/D layout: row = vgpr + 8*(lane>>4)
        int n = n0 + 16 * ni + l15;              //             col = lane&15
        float val = acc[mi][ni][r] + bias[n];
        int bi = m >> 11, t = m & (T_ - 1);
        if (n < C_) {
          int h = n >> 6, d = n & 63;
          Qb[(((size_t)bi * H_ + h) * T_ + t) * D_ + d] = (__bf16)val;
        } else if (n < 2 * C_) {
          int nn = n - C_, h = nn >> 6, d = nn & 63;
          Kb[(((size_t)bi * H_ + h) * T_ + t) * D_ + d] = (__bf16)val;
        } else {
          int nn = n - 2 * C_, h = nn >> 6, d = nn & 63;
          Vt[(((size_t)bi * H_ + h) * D_ + d) * T_ + t] = (__bf16)val;   // transposed
        }
      }
}

// ---------------- attention: one block per (b, h, 16-row q tile) ----------------
#define SLD 2052   // f32 score row stride (padded, 16B aligned)
#define PLD 2056   // bf16 prob  row stride (padded, 16B aligned)
#define ATTN_SMEM (16 * SLD * 4 + 16 * PLD * 2)

__global__ __launch_bounds__(256) void k_attn(
    const __bf16* __restrict__ Qb, const __bf16* __restrict__ Kb,
    const __bf16* __restrict__ Vt, __bf16* __restrict__ Ab) {
  extern __shared__ char smem[];
  float*  S = (float*)smem;                        // [16][SLD] f32 scores
  __bf16* P = (__bf16*)(smem + 16 * SLD * 4);      // [16][PLD] bf16 exp(s-m)
  __shared__ float rowsum[16];

  const int qt = blockIdx.x, h = blockIdx.y, bb = blockIdx.z;
  const int bh = bb * H_ + h;
  const int q0 = qt * 16;
  const int kvlen = (qt + 1) * 16;                 // causal: keys 0..q0+15
  const int kvpad = (kvlen + 31) & ~31;            // round to WMMA K step
  const int tid = threadIdx.x, lane = tid & 31, wv = tid >> 5;
  const int half = lane >> 4, l15 = lane & 15;
  const int koffA = half * 8, koffB = half * 8;
  const float scale = 0.125f;                      // 1/sqrt(64)
  const v8f vzero = {0.f,0.f,0.f,0.f,0.f,0.f,0.f,0.f};

  // ---- phase 1: S = mask(scale * Q Kᵀ), 8 waves over key tiles ----
  Frag qa0, qa1;
  const __bf16* qrow = Qb + ((size_t)bh * T_ + q0 + l15) * D_;
  load_a(qa0, qrow, 0, koffA);
  load_a(qa1, qrow, 32, koffA);
  const int ntiles = qt + 1;
  for (int kt = wv; kt < ntiles; kt += 8) {
    const int jb = kt * 16;
    const __bf16* krow = Kb + ((size_t)bh * T_ + jb + l15) * D_;  // B col n = K row n
    Frag kb0, kb1;
    load_b(kb0, krow, 0, koffB);
    load_b(kb1, krow, 32, koffB);
    v8f acc = vzero;
    acc = wmma_bf16(qa0.v, kb0.v, acc);
    acc = wmma_bf16(qa1.v, kb1.v, acc);
    for (int r = 0; r < 8; ++r) {
      int mloc = r + 8 * half;
      int key = jb + l15;
      float v = acc[r] * scale;
      S[mloc * SLD + key] = (key <= q0 + mloc) ? v : -1e30f;
    }
  }
  __syncthreads();

  // ---- phase 2: row softmax (unnormalized), 16 threads per row ----
  {
    int row = tid >> 4, sub = tid & 15;            // rows 2w,2w+1 live in wave w halves
    const float* Srow = S + row * SLD;
    __bf16* Prow = P + row * PLD;
    float m = -1e30f;
    for (int c = sub; c < kvlen; c += 16) m = fmaxf(m, Srow[c]);
    for (int o = 8; o; o >>= 1) m = fmaxf(m, __shfl_xor(m, o, 32));  // xor<16 stays in half
    float s = 0.f;
    for (int c = sub; c < kvlen; c += 16) {
      float e = __expf(Srow[c] - m);
      s += e;
      Prow[c] = (__bf16)e;
    }
    for (int c = kvlen + sub; c < kvpad; c += 16) Prow[c] = (__bf16)0.f;  // zero pad
    for (int o = 8; o; o >>= 1) s += __shfl_xor(s, o, 32);
    if (sub == 0) rowsum[row] = s;
  }
  __syncthreads();

  // ---- phase 3: O = (P V) / rowsum, waves 0..3 each own a 16-wide d tile ----
  if (wv < 4) {
    const __bf16* vrow = Vt + ((size_t)bh * D_ + wv * 16 + l15) * T_;  // B col n = Vt row
    const __bf16* prow = P + l15 * PLD;                                 // A row = lane&15
    v8f acc = vzero;
    for (int jb = 0; jb < kvpad; jb += 32) {
      Frag pa, vb;
      load_a(pa, prow, jb, koffA);    // ds_load_b128 from LDS
      load_b(vb, vrow, jb, koffB);
      acc = wmma_bf16(pa.v, vb.v, acc);
    }
    for (int r = 0; r < 8; ++r) {
      int mloc = r + 8 * half;
      int d = wv * 16 + l15;
      float v = acc[r] / rowsum[mloc];
      Ab[((size_t)bb * T_ + q0 + mloc) * C_ + h * D_ + d] = (__bf16)v;
    }
  }
}

// ---------------- output projection: out = Ab[8192,1024] @ Wprojᵀ + bias (f32) ----------------
__global__ __launch_bounds__(256) void k_gemm_proj(
    const __bf16* __restrict__ Ab, const __bf16* __restrict__ Wt,
    const float* __restrict__ bias, float* __restrict__ out) {
  const int tid = threadIdx.x;
  const int lane = tid & 31, wv = tid >> 5;
  const int half = lane >> 4, l15 = lane & 15;
  const int koffA = half * 8, koffB = half * 8;
  const int m0 = blockIdx.y * 64 + (wv >> 2) * 32;
  const int n0 = blockIdx.x * 256 + (wv & 3) * 64;

  const v8f vzero = {0.f,0.f,0.f,0.f,0.f,0.f,0.f,0.f};
  v8f acc[2][4];
  for (int i = 0; i < 2; ++i)
    for (int j = 0; j < 4; ++j) acc[i][j] = vzero;

  const __bf16* arow0 = Ab + (size_t)(m0 + l15) * C_;
  const __bf16* arow1 = Ab + (size_t)(m0 + 16 + l15) * C_;
  const __bf16* brow[4];
  for (int j = 0; j < 4; ++j) brow[j] = Wt + (size_t)(n0 + 16 * j + l15) * C_;

  for (int kk = 0; kk < C_; kk += 32) {
    Frag a0, a1, bf[4];
    load_a(a0, arow0, kk, koffA);
    load_a(a1, arow1, kk, koffA);
    for (int j = 0; j < 4; ++j) load_b(bf[j], brow[j], kk, koffB);
    for (int j = 0; j < 4; ++j) {
      acc[0][j] = wmma_bf16(a0.v, bf[j].v, acc[0][j]);
      acc[1][j] = wmma_bf16(a1.v, bf[j].v, acc[1][j]);
    }
  }

  for (int mi = 0; mi < 2; ++mi)
    for (int ni = 0; ni < 4; ++ni)
      for (int r = 0; r < 8; ++r) {
        int m = m0 + 16 * mi + r + 8 * half;
        int n = n0 + 16 * ni + l15;
        out[(size_t)m * C_ + n] = acc[mi][ni][r] + bias[n];
      }
}

// ---------------- launch ----------------
extern "C" void kernel_launch(void* const* d_in, const int* in_sizes, int n_in,
                              void* d_out, int out_size, void* d_ws, size_t ws_size,
                              hipStream_t stream) {
  (void)in_sizes; (void)n_in; (void)out_size; (void)ws_size;
  const float* X     = (const float*)d_in[0];  // [B,T,C]
  const float* Wqkv  = (const float*)d_in[1];  // [C,3C]
  const float* bqkv  = (const float*)d_in[2];  // [3C]
  const float* Wproj = (const float*)d_in[3];  // [C,C]
  const float* bproj = (const float*)d_in[4];  // [C]
  float* out = (float*)d_out;                  // [B,T,C]

  char* ws = (char*)d_ws;
  const size_t nX = (size_t)B_ * T_ * C_;          // 8,388,608
  size_t off = 0;
  __bf16* Xb     = (__bf16*)(ws + off); off += nX * 2;
  __bf16* WqkvT  = (__bf16*)(ws + off); off += (size_t)C_ * C3_ * 2;
  __bf16* WprojT = (__bf16*)(ws + off); off += (size_t)C_ * C_ * 2;
  __bf16* Qb     = (__bf16*)(ws + off); off += nX * 2;
  __bf16* Kb     = (__bf16*)(ws + off); off += nX * 2;
  __bf16* Vt     = (__bf16*)(ws + off); off += nX * 2;
  __bf16* Ab     = (__bf16*)(ws + off); off += nX * 2;   // ~92 MB total

  (void)hipFuncSetAttribute((const void*)k_attn,
                            hipFuncAttributeMaxDynamicSharedMemorySize, ATTN_SMEM);

  k_cvt<<<4096, 256, 0, stream>>>(X, Xb, (int)nX);
  k_transpose<<<2048, 256, 0, stream>>>(Wqkv, WqkvT, C_, C3_);
  k_transpose<<<1024, 256, 0, stream>>>(Wproj, WprojT, C_, C_);

  k_gemm_qkv<<<dim3(C3_ / 256, (B_ * T_) / 64), 256, 0, stream>>>(
      Xb, WqkvT, bqkv, Qb, Kb, Vt);

  k_attn<<<dim3(T_ / 16, H_, B_), 256, ATTN_SMEM, stream>>>(Qb, Kb, Vt, Ab);

  k_gemm_proj<<<dim3(C_ / 256, (B_ * T_) / 64), 256, 0, stream>>>(
      Ab, WprojT, bproj, out);
}
